// GCNNet_10694468567328
// MI455X (gfx1250) — matrix operand order repaired and checked by reading, not compiled, not run on hardware
//
#include <hip/hip_runtime.h>
#include <hip/hip_bf16.h>

typedef __attribute__((ext_vector_type(16))) _Float16 v16h;
typedef __attribute__((ext_vector_type(8)))  float    v8f;

#define F1 128
#define F2 64

// ---------------- degree / normalization ----------------
__global__ void k_fill_deg(float* deg, int n) {
  int i = blockIdx.x * blockDim.x + threadIdx.x;
  if (i < n) deg[i] = 1.0f;  // self-loop
}

__global__ void k_deg_scatter(const int* __restrict__ col, float* deg, int E) {
  int e = blockIdx.x * blockDim.x + threadIdx.x;
  if (e < E) atomicAdd(&deg[col[e]], 1.0f);
}

__global__ void k_dinv(float* d, int n) {
  int i = blockIdx.x * blockDim.x + threadIdx.x;
  if (i < n) d[i] = rsqrtf(d[i]);
}

// ---------------- layer-1 scalar aggregation ----------------
__global__ void k_agg_init(const float* __restrict__ x, const float* __restrict__ dinv,
                           float* agg, int n) {
  int i = blockIdx.x * blockDim.x + threadIdx.x;
  if (i < n) agg[i] = dinv[i] * dinv[i] * x[i];
}

__global__ void k_agg_scatter(const int* __restrict__ row, const int* __restrict__ col,
                              const float* __restrict__ x, const float* __restrict__ dinv,
                              float* agg, float* nrm, int E) {
  int e = blockIdx.x * blockDim.x + threadIdx.x;
  if (e < E) {
    int r = row[e], c = col[e];
    float w = dinv[r] * dinv[c];
    nrm[e] = w;
    atomicAdd(&agg[c], w * x[r]);
  }
}

// ---------------- WMMA feature transform: hw = relu(agg*W1+b1) @ W2 ----------------
// One wave handles a 16-node tile: H[16x128](f16) x W2[128x64](f16) -> hw[16x64](f32)
__global__ __launch_bounds__(256) void k_feat_wmma(
    const float* __restrict__ agg, const float* __restrict__ W1,
    const float* __restrict__ b1, const float* __restrict__ W2,
    float* __restrict__ hw, int n) {
  int lane = threadIdx.x & 31;
  int wave = threadIdx.x >> 5;
  int m0 = (blockIdx.x * 8 + wave) * 16;
  if (m0 >= n) return;               // wave-uniform: EXEC stays all-ones

  int mrow = lane & 15;              // M row (A) / N column (B,C) within tile
  int hi   = lane >> 4;              // half-wave selector
  float s = agg[m0 + mrow];

  // A fragments: 16-bit A 16x32 layout. lanes0-15: halfs[0..7]=K+0..7, [8..15]=K+16..23
  //              lanes16-31: halfs[0..7]=K+8..15, [8..15]=K+24..31
  v16h a[4];
#pragma unroll
  for (int kc = 0; kc < 4; ++kc) {
    int kb = kc * 32 + hi * 8;
#pragma unroll
    for (int j = 0; j < 8; ++j) {
      int k0 = kb + j;
      int k1 = kb + 16 + j;
      float h0 = s * W1[k0] + b1[k0]; h0 = h0 > 0.0f ? h0 : 0.0f;
      float h1 = s * W1[k1] + b1[k1]; h1 = h1 > 0.0f ? h1 : 0.0f;
      a[kc][j]     = (_Float16)h0;
      a[kc][j + 8] = (_Float16)h1;
    }
  }

  // B fragments: 32x16 tile of W2; lanes0-15 hold K=kc*32+{0..7,16..23} of col N,
  //              lanes16-31 hold K=kc*32+{8..15,24..31} of col N.
#pragma unroll
  for (int nt = 0; nt < 4; ++nt) {
    int ncol = nt * 16 + mrow;
    v8f c = {};
#pragma unroll
    for (int kc = 0; kc < 4; ++kc) {
      v16h b;
      int kb = kc * 32 + hi * 8;
#pragma unroll
      for (int j = 0; j < 16; ++j)
        b[j] = (_Float16)W2[(kb + j + (j >= 8 ? 8 : 0)) * F2 + ncol];
      c = __builtin_amdgcn_wmma_f32_16x16x32_f16(false, a[kc], false, b,
                                                 (short)0, c, false, false);
    }
    // C layout: lanes0-15 -> rows M=0..7 in v0..7, lanes16-31 -> rows M=8..15
#pragma unroll
    for (int v = 0; v < 8; ++v)
      hw[(size_t)(m0 + hi * 8 + v) * F2 + ncol] = c[v];
  }
}

// ---------------- layer-2: z init (self-loop + bias) and 64-wide scatter ----------------
__global__ void k_z_init(const float* __restrict__ hw, const float* __restrict__ dinv,
                         const float* __restrict__ b2, float* z, int n) {
  int tid = blockIdx.x * blockDim.x + threadIdx.x;
  int i = tid >> 6, f = tid & 63;
  if (i < n) {
    float d = dinv[i];
    z[(size_t)i * F2 + f] = d * d * hw[(size_t)i * F2 + f] + b2[f];
  }
}

__global__ void k_z_scatter(const int* __restrict__ row, const int* __restrict__ col,
                            const float* __restrict__ nrm, const float* __restrict__ hw,
                            float* z, int E) {
  int tid = blockIdx.x * blockDim.x + threadIdx.x;
  int e = tid >> 6, f = tid & 63;
  if (e < E)
    atomicAdd(&z[(size_t)col[e] * F2 + f], nrm[e] * hw[(size_t)row[e] * F2 + f]);
}

// ---------------- decode: logits = dot(z[a], z[b]) over pos then neg edges ----------------
__global__ void k_decode(const int* __restrict__ pos, const int* __restrict__ neg,
                         const float* __restrict__ z, float* out, int E) {
  int e = blockIdx.x * blockDim.x + threadIdx.x;
  if (e >= 2 * E) return;
  int ia, ib;
  if (e < E) { ia = pos[e];     ib = pos[E + e]; }
  else       { ia = neg[e - E]; ib = neg[E + e - E]; }
  const float4* za = (const float4*)(z + (size_t)ia * F2);
  const float4* zb = (const float4*)(z + (size_t)ib * F2);
  float acc = 0.0f;
#pragma unroll
  for (int j = 0; j < 16; ++j) {
    float4 p = za[j], q = zb[j];
    acc += p.x * q.x + p.y * q.y + p.z * q.z + p.w * q.w;
  }
  out[e] = acc;
}

extern "C" void kernel_launch(void* const* d_in, const int* in_sizes, int n_in,
                              void* d_out, int out_size, void* d_ws, size_t ws_size,
                              hipStream_t stream) {
  const float* x   = (const float*)d_in[0];
  const int*   ei  = (const int*)d_in[1];
  const int*   nei = (const int*)d_in[2];
  const float* W1  = (const float*)d_in[3];
  const float* b1  = (const float*)d_in[4];
  const float* W2  = (const float*)d_in[5];
  const float* b2  = (const float*)d_in[6];
  float* out = (float*)d_out;

  int n = in_sizes[0];       // x is [N,1]
  int E = in_sizes[1] / 2;   // edge_index is [2,E]
  const int* row = ei;
  const int* col = ei + E;

  // workspace carve-up (floats): dinv[n] | agg[n] | nrm[E] | hw[n*64] | z[n*64]
  float* ws   = (float*)d_ws;
  float* dinv = ws;
  float* agg  = ws + n;
  float* nrm  = ws + 2 * (size_t)n;
  float* hw   = nrm + E;
  float* z    = hw + (size_t)n * F2;

  const int B = 256;
  int gn  = (n + B - 1) / B;
  int gE  = (E + B - 1) / B;
  int gnf = (int)(((size_t)n * F2 + B - 1) / B);
  int gEf = (int)(((size_t)E * F2 + B - 1) / B);
  int g2E = (2 * E + B - 1) / B;
  int gW  = (n + 127) / 128;   // 8 waves/block, 16 nodes/wave

  k_fill_deg   <<<gn,  B, 0, stream>>>(dinv, n);
  k_deg_scatter<<<gE,  B, 0, stream>>>(col, dinv, E);
  k_dinv       <<<gn,  B, 0, stream>>>(dinv, n);
  k_agg_init   <<<gn,  B, 0, stream>>>(x, dinv, agg, n);
  k_agg_scatter<<<gE,  B, 0, stream>>>(row, col, x, dinv, agg, nrm, E);
  k_feat_wmma  <<<gW,  B, 0, stream>>>(agg, W1, b1, W2, hw, n);
  k_z_init     <<<gnf, B, 0, stream>>>(hw, dinv, b2, z, n);
  k_z_scatter  <<<gEf, B, 0, stream>>>(row, col, nrm, hw, z, E);
  k_decode     <<<g2E, B, 0, stream>>>(ei, nei, z, out, E);
}
